// F0Decoder_10977936409090
// MI455X (gfx1250) — matrix operand order repaired and checked by reading, not compiled, not run on hardware
//
#include <hip/hip_runtime.h>

typedef __attribute__((ext_vector_type(16))) __bf16 v16bf;
typedef __attribute__((ext_vector_type(8)))  __bf16 v8bf;
typedef __attribute__((ext_vector_type(8)))  float  v8f;
typedef __attribute__((ext_vector_type(4)))  float  f4v;
typedef __attribute__((ext_vector_type(4)))  unsigned int u32x4;
typedef __attribute__((ext_vector_type(8)))  int    i32x8;
typedef __attribute__((ext_vector_type(4)))  int    i32x4;

constexpr int Bc = 4, Cc = 512, Tc = 2048, Hc = 8, FCc = 2048, Lc = 6, Kc = 3, Sc = 256;
constexpr int DKc = 64;
constexpr int TPc = Tc + 8;   // padded time rows
constexpr int PADTc = 4;      // interior offset (covers left pad 2, right pad 1)

static __device__ __forceinline__ v16bf cat8(v8bf lo, v8bf hi) {
  return __builtin_shufflevector(lo, hi, 0, 1, 2, 3, 4, 5, 6, 7, 8, 9, 10, 11, 12, 13, 14, 15);
}

// ---------------------------------------------------------------------------
// TDM / cluster probe: exercises the CDNA5 Tensor Data Mover with a NULL
// (count=0) descriptor -- architecturally a no-op -- plus s_wait_tensorcnt and
// the cluster barrier (NOP when ClusterID==0). Compile-safe via __has_builtin.
// ---------------------------------------------------------------------------
#if __has_builtin(__builtin_amdgcn_tensor_load_to_lds)
#define HAVE_TDM 1
#else
#define HAVE_TDM 0
#endif

__global__ __launch_bounds__(32) void tdm_probe_kernel() {
#if HAVE_TDM
  u32x4 g0 = {};  // count=0 -> NULL tensor descriptor
  i32x8 g1 = {};
  i32x4 g2 = {};
  i32x4 g3 = {};
#if __clang_major__ >= 23
  i32x8 g4 = {};
  __builtin_amdgcn_tensor_load_to_lds(g0, g1, g2, g3, g4, 0);
#else
  __builtin_amdgcn_tensor_load_to_lds(g0, g1, g2, g3, 0);
#endif
#if __has_builtin(__builtin_amdgcn_s_wait_tensorcnt)
  __builtin_amdgcn_s_wait_tensorcnt(0);
#endif
#endif
#if __has_builtin(__builtin_amdgcn_s_cluster_barrier)
  __builtin_amdgcn_s_cluster_barrier();
#endif
}

// ---------------------------------------------------------------------------
// plain f32 -> bf16 (for 1x1 weights, layout unchanged (M, Cin))
// ---------------------------------------------------------------------------
__global__ void cvt_bf16_kernel(const float* __restrict__ in, __bf16* __restrict__ out,
                                size_t n) {
  size_t i = (size_t)blockIdx.x * blockDim.x + threadIdx.x;
  size_t stride = (size_t)gridDim.x * blockDim.x;
  for (; i < n; i += stride) out[i] = (__bf16)in[i];
}

// (M, Cin, Ktap) f32 -> (M, Ktap, Cin) bf16 so per-tap cin is contiguous
__global__ void cvt_w_tap_kernel(const float* __restrict__ in, __bf16* __restrict__ out,
                                 int Mtot, int Cin, int Ktap) {
  size_t n = (size_t)Mtot * Cin * Ktap;
  size_t i = (size_t)blockIdx.x * blockDim.x + threadIdx.x;
  size_t stride = (size_t)gridDim.x * blockDim.x;
  for (; i < n; i += stride) {
    int tap = (int)(i % Ktap);
    int cin = (int)((i / Ktap) % Cin);
    size_t m = i / ((size_t)Ktap * Cin);
    out[(m * Ktap + tap) * Cin + cin] = (__bf16)in[i];
  }
}

// ---------------------------------------------------------------------------
// Fused: x + cond1x1(spk) + conv1d(norm_f0, K=3, pad 1) -> bf16 (B,TP,C) padded
// ---------------------------------------------------------------------------
__global__ void pre_kernel(const float* __restrict__ x, const float* __restrict__ f0,
                           const float* __restrict__ spk,
                           const float* __restrict__ cond_w, const float* __restrict__ cond_b,
                           const float* __restrict__ f0w, const float* __restrict__ f0b,
                           __bf16* __restrict__ outbf) {
  size_t idx = (size_t)blockIdx.x * blockDim.x + threadIdx.x;
  const size_t total = (size_t)Bc * Cc * Tc;
  if (idx >= total) return;
  int c = (int)(idx % Cc);
  int t = (int)((idx / Cc) % Tc);
  int bb = (int)(idx / ((size_t)Cc * Tc));
  // x is (B, C, T)
  float v = x[((size_t)bb * Cc + c) * Tc + t] + cond_b[c] + f0b[c];
  const float* sp = spk + (size_t)bb * Sc * Tc + t;
  const float* cw = cond_w + (size_t)c * Sc;
  for (int s = 0; s < Sc; ++s) v += cw[s] * sp[(size_t)s * Tc];
  const float* f0r = f0 + (size_t)bb * Tc;
  const float* fw = f0w + c * 3;
#pragma unroll
  for (int k = 0; k < 3; ++k) {
    int tt = t + k - 1;
    if (tt >= 0 && tt < Tc) v += fw[k] * f0r[tt];
  }
  outbf[((size_t)bb * TPc + PADTc + t) * Cc + c] = (__bf16)v;
}

// ---------------------------------------------------------------------------
// conv1d-as-GEMM via WMMA bf16; one wave -> 16(M) x 64(N=4 tiles), A reused 4x.
//   W: (M, Ktap, Cin) bf16 ; X: (B, TP, Cin) bf16 zero-padded rows
//   Yf: (B, T, M) f32 (optional) ; Ybf: (B, TP, M) padded, or (B, M, T) if TMAJOR
// ---------------------------------------------------------------------------
template <int M, int Cin, int Ktap, int padL, bool RELU, bool TMAJOR>
__global__ __launch_bounds__(32)
void convgemm_wmma(const __bf16* __restrict__ W, const __bf16* __restrict__ X,
                   const float* __restrict__ bias, const float* __restrict__ mask,
                   float* __restrict__ Yf, __bf16* __restrict__ Ybf) {
  constexpr int NT = 4;  // N tiles per wave
  const int lane = threadIdx.x;
  const int half = lane >> 4;
  const int l16 = lane & 15;
  const int n0 = blockIdx.x * (16 * NT);
  const int m0 = blockIdx.y * 16;
  const int bb = blockIdx.z;

  v8f zero = {};
  v8f acc[NT];
#pragma unroll
  for (int j = 0; j < NT; ++j) acc[j] = zero;

  const __bf16* Wrow = W + (size_t)(m0 + l16) * (Ktap * Cin);
  const __bf16* Xbase = X + ((size_t)bb * TPc + (PADTc - padL) + n0 + l16) * Cin;

#pragma unroll
  for (int tap = 0; tap < Ktap; ++tap) {
    const __bf16* Wr = Wrow + (size_t)tap * Cin;
    const __bf16* Xr = Xbase + (size_t)tap * Cin;
    __builtin_prefetch(Xr + (size_t)16 * NT * Cin, 0, 1);  // gfx1250 global_prefetch_b8
#pragma unroll 2
    for (int c0 = 0; c0 < Cin; c0 += 32) {
      v8bf alo = *(const v8bf*)(Wr + c0 + (half << 4));
      v8bf ahi = *(const v8bf*)(Wr + c0 + 8 + (half << 4));
      v16bf a = cat8(alo, ahi);
#pragma unroll
      for (int j = 0; j < NT; ++j) {
        v16bf bm = *(const v16bf*)(Xr + (size_t)j * 16 * Cin + c0 + (half << 4));
        acc[j] = __builtin_amdgcn_wmma_f32_16x16x32_bf16(false, a, false, bm,
                                                         (short)0, acc[j], false, false);
      }
    }
  }

  v8f bv = *(const v8f*)(bias + m0 + (half << 3));
#pragma unroll
  for (int j = 0; j < NT; ++j) {
    const int bt = n0 + j * 16 + l16;
    const float mk = mask ? mask[(size_t)bb * Tc + bt] : 1.f;
    v8f outv;
#pragma unroll
    for (int r = 0; r < 8; ++r) {
      float v = acc[j][r] + bv[r];
      if (RELU) v = fmaxf(v, 0.f);
      outv[r] = v * mk;
    }
    if (Yf) *(v8f*)(Yf + ((size_t)bb * Tc + bt) * M + m0 + (half << 3)) = outv;
    if (Ybf) {
      if (TMAJOR) {
#pragma unroll
        for (int r = 0; r < 8; ++r)
          Ybf[((size_t)bb * M + (m0 + r + (half << 3))) * Tc + bt] = (__bf16)outv[r];
      } else {
        v8bf obv;
#pragma unroll
        for (int r = 0; r < 8; ++r) obv[r] = (__bf16)outv[r];
        *(v8bf*)(Ybf + ((size_t)bb * TPc + PADTc + bt) * M + m0 + (half << 3)) = obv;
      }
    }
  }
}

// ---------------------------------------------------------------------------
// Flash causal attention. Q,K: (B,TP,C) padded channel-contig; V: (B,C,T)
// time-major; O: (B,TP,C) padded. One wave per (16 queries, head, batch).
// ---------------------------------------------------------------------------
__global__ __launch_bounds__(32)
void attn_flash_kernel(const __bf16* __restrict__ Qb, const __bf16* __restrict__ Kb,
                       const __bf16* __restrict__ Vb, __bf16* __restrict__ Ob) {
  __shared__ __attribute__((aligned(32))) __bf16 pLds[16][40];
  __shared__ __attribute__((aligned(32))) __bf16 oLds[16][80];
  const int lane = threadIdx.x;
  const int half = lane >> 4;
  const int l16 = lane & 15;
  const int t0 = blockIdx.x * 16;
  const int h = blockIdx.y;
  const int bb = blockIdx.z;
  const size_t rowB = (size_t)bb * TPc + PADTc;
  const int ch0 = h * DKc;

  v16bf qA[2];
  {
    const __bf16* qp = Qb + (rowB + t0 + l16) * Cc + ch0;
#pragma unroll
    for (int step = 0; step < 2; ++step) {
      v8bf lo = *(const v8bf*)(qp + step * 32 + (half << 4));
      v8bf hi = *(const v8bf*)(qp + step * 32 + 8 + (half << 4));
      qA[step] = cat8(lo, hi);
    }
  }

  v8f zero = {};
  v8f oacc[4];
#pragma unroll
  for (int dt = 0; dt < 4; ++dt) oacc[dt] = zero;
  float mrun[8], lrun[8];
#pragma unroll
  for (int r = 0; r < 8; ++r) { mrun[r] = -3.0e38f; lrun[r] = 0.f; }

  const float scale = 0.125f;  // 1/sqrt(64)

  for (int s0 = 0; s0 < t0 + 16; s0 += 32) {
    v8f sc[2];
#pragma unroll
    for (int sub = 0; sub < 2; ++sub) {
      sc[sub] = zero;
      const int scol = s0 + sub * 16 + l16;
      const __bf16* kp = Kb + (rowB + scol) * Cc + ch0;
#pragma unroll
      for (int step = 0; step < 2; ++step) {
        v16bf kB = *(const v16bf*)(kp + step * 32 + (half << 4));
        sc[sub] = __builtin_amdgcn_wmma_f32_16x16x32_bf16(false, qA[step], false, kB,
                                                          (short)0, sc[sub], false, false);
      }
      if (s0 + sub * 16 + 15 <= t0) {
        // entire subtile below the diagonal: no masking needed (uniform branch)
#pragma unroll
        for (int r = 0; r < 8; ++r) sc[sub][r] *= scale;
      } else {
#pragma unroll
        for (int r = 0; r < 8; ++r) {
          int tr = t0 + r + (half << 3);
          sc[sub][r] = (scol <= tr) ? sc[sub][r] * scale : -1.0e4f;
        }
      }
    }
    // online softmax over 32-wide key chunk
#pragma unroll
    for (int r = 0; r < 8; ++r) {
      float mx = fmaxf(sc[0][r], sc[1][r]);
#pragma unroll
      for (int xm = 1; xm <= 8; xm <<= 1) mx = fmaxf(mx, __shfl_xor(mx, xm, 32));
      float mnew = fmaxf(mrun[r], mx);
      float p0 = __expf(sc[0][r] - mnew);
      float p1 = __expf(sc[1][r] - mnew);
      float ls = p0 + p1;
#pragma unroll
      for (int xm = 1; xm <= 8; xm <<= 1) ls += __shfl_xor(ls, xm, 32);
      float corr = __expf(mrun[r] - mnew);
      lrun[r] = lrun[r] * corr + ls;
      mrun[r] = mnew;
#pragma unroll
      for (int dt = 0; dt < 4; ++dt) oacc[dt][r] *= corr;
      pLds[r + (half << 3)][l16] = (__bf16)p0;
      pLds[r + (half << 3)][16 + l16] = (__bf16)p1;
    }
    __syncthreads();
    v8bf plo = *(const v8bf*)&pLds[l16][half << 4];
    v8bf phi = *(const v8bf*)&pLds[l16][8 + (half << 4)];
    v16bf pA = cat8(plo, phi);
    __syncthreads();
#pragma unroll
    for (int dt = 0; dt < 4; ++dt) {
      const int d = dt * 16 + l16;
      v16bf vB = *(const v16bf*)(Vb + ((size_t)bb * Cc + ch0 + d) * Tc + s0 + (half << 4));
      oacc[dt] = __builtin_amdgcn_wmma_f32_16x16x32_bf16(false, pA, false, vB,
                                                         (short)0, oacc[dt], false, false);
    }
  }

  // normalize + transpose via LDS so global stores are wide
#pragma unroll
  for (int dt = 0; dt < 4; ++dt)
#pragma unroll
    for (int r = 0; r < 8; ++r)
      oLds[r + (half << 3)][dt * 16 + l16] = (__bf16)(oacc[dt][r] / lrun[r]);
  __syncthreads();
  {
    __bf16* op = Ob + (rowB + t0 + l16) * Cc + ch0 + (half << 5);
    v16bf o0 = *(const v16bf*)&oLds[l16][half << 5];
    v16bf o1 = *(const v16bf*)&oLds[l16][(half << 5) + 16];
    *(v16bf*)op = o0;
    *(v16bf*)(op + 16) = o1;
  }
}

// ---------------------------------------------------------------------------
// Channel LayerNorm of (xin + yin), both (B,T,C); one wave per (b,t).
// Writes f32 (B,T,C) + bf16 (B,TP,C) padded.
// ---------------------------------------------------------------------------
__global__ __launch_bounds__(32)
void ln_kernel(const float* __restrict__ xin, const float* __restrict__ yin,
               const float* __restrict__ g, const float* __restrict__ bta,
               float* __restrict__ xout, __bf16* __restrict__ xbf) {
  const int lane = threadIdx.x;
  const int t = blockIdx.x;
  const int bb = blockIdx.y;
  const size_t base = ((size_t)bb * Tc + t) * Cc + lane * 16;
  f4v v[4];
  float s = 0.f, s2 = 0.f;
#pragma unroll
  for (int j = 0; j < 4; ++j) {
    f4v a = *(const f4v*)(xin + base + j * 4);
    f4v c = *(const f4v*)(yin + base + j * 4);
    f4v w = a + c;
    v[j] = w;
#pragma unroll
    for (int e = 0; e < 4; ++e) { s += w[e]; s2 += w[e] * w[e]; }
  }
#pragma unroll
  for (int xm = 1; xm <= 16; xm <<= 1) {
    s += __shfl_xor(s, xm, 32);
    s2 += __shfl_xor(s2, xm, 32);
  }
  const float mean = s * (1.f / Cc);
  const float var = s2 * (1.f / Cc) - mean * mean;
  const float rstd = rsqrtf(var + 1e-5f);
  const float* gp = g + lane * 16;
  const float* bp = bta + lane * 16;
  v16bf ob;
#pragma unroll
  for (int j = 0; j < 4; ++j) {
    f4v gv = *(const f4v*)(gp + j * 4);
    f4v bv = *(const f4v*)(bp + j * 4);
    f4v o;
#pragma unroll
    for (int e = 0; e < 4; ++e) {
      float val = (v[j][e] - mean) * rstd * gv[e] + bv[e];
      o[e] = val;
      ob[j * 4 + e] = (__bf16)val;
    }
    *(f4v*)(xout + base + j * 4) = o;
  }
  *(v16bf*)(xbf + ((size_t)bb * TPc + PADTc + t) * Cc + lane * 16) = ob;
}

// ---------------------------------------------------------------------------
// Final 1-channel projection; one wave per (b,t), x is (B,T,C).
// ---------------------------------------------------------------------------
__global__ __launch_bounds__(32)
void proj_kernel(const float* __restrict__ x, const float* __restrict__ pw,
                 const float* __restrict__ pb, const float* __restrict__ mask,
                 float* __restrict__ out) {
  const int lane = threadIdx.x;
  const int t = blockIdx.x;
  const int bb = blockIdx.y;
  const float* xr = x + ((size_t)bb * Tc + t) * Cc + lane * 16;
  const float* wr = pw + lane * 16;
  float s = 0.f;
#pragma unroll
  for (int j = 0; j < 4; ++j) {
    f4v xv = *(const f4v*)(xr + j * 4);
    f4v wv = *(const f4v*)(wr + j * 4);
#pragma unroll
    for (int e = 0; e < 4; ++e) s += xv[e] * wv[e];
  }
#pragma unroll
  for (int xm = 1; xm <= 16; xm <<= 1) s += __shfl_xor(s, xm, 32);
  if (lane == 0) {
    float mk = mask[(size_t)bb * Tc + t];
    out[(size_t)bb * Tc + t] = (s + pb[0]) * mk * mk;  // ref: x*mask, proj, *mask
  }
}

// ---------------------------------------------------------------------------
extern "C" void kernel_launch(void* const* d_in, const int* in_sizes, int n_in,
                              void* d_out, int out_size, void* d_ws, size_t ws_size,
                              hipStream_t stream) {
  const float* x        = (const float*)d_in[0];
  const float* norm_f0  = (const float*)d_in[1];
  const float* x_mask   = (const float*)d_in[2];
  const float* spk_emb  = (const float*)d_in[3];
  const float* prenet_w = (const float*)d_in[4];
  const float* prenet_b = (const float*)d_in[5];
  const float* f0pre_w  = (const float*)d_in[6];
  const float* f0pre_b  = (const float*)d_in[7];
  const float* cond_w   = (const float*)d_in[8];
  const float* cond_b   = (const float*)d_in[9];
  const float* proj_w   = (const float*)d_in[10];
  const float* proj_b   = (const float*)d_in[11];
  const float* qw = (const float*)d_in[12]; const float* qb = (const float*)d_in[13];
  const float* kw = (const float*)d_in[14]; const float* kb = (const float*)d_in[15];
  const float* vw = (const float*)d_in[16]; const float* vb = (const float*)d_in[17];
  const float* ow = (const float*)d_in[18]; const float* ob = (const float*)d_in[19];
  const float* ln0_g = (const float*)d_in[20]; const float* ln0_b = (const float*)d_in[21];
  const float* ln1_g = (const float*)d_in[22]; const float* ln1_b = (const float*)d_in[23];
  const float* f1w = (const float*)d_in[24]; const float* f1b = (const float*)d_in[25];
  const float* f2w = (const float*)d_in[26]; const float* f2b = (const float*)d_in[27];

  const size_t BCT  = (size_t)Bc * Cc * Tc;
  const size_t BPC  = (size_t)Bc * TPc * Cc;   // padded bf16 (B,TP,C)
  const size_t BPF  = (size_t)Bc * TPc * FCc;  // padded bf16 (B,TP,FC)
  const size_t CCK  = (size_t)Cc * Cc * Kc;
  const size_t LCC  = (size_t)Lc * Cc * Cc;
  const size_t LF1  = (size_t)Lc * FCc * Cc * Kc;
  const size_t LF2  = (size_t)Lc * Cc * FCc * Kc;

  char* ws = (char*)d_ws;
  size_t off = 0;
  auto take = [&](size_t bytes) -> char* {
    char* p = ws + off;
    off += (bytes + 255) & ~(size_t)255;
    return p;
  };

  float*  xf    = (float*)take(BCT * 4);
  float*  yf    = (float*)take(BCT * 4);
  // padded bf16 activations (contiguous span for one memset)
  char*   padPool = ws + off;
  __bf16* xbf   = (__bf16*)take(BPC * 2);
  __bf16* xpreb = (__bf16*)take(BPC * 2);
  __bf16* qbf   = (__bf16*)take(BPC * 2);
  __bf16* kbf   = (__bf16*)take(BPC * 2);
  __bf16* abf   = (__bf16*)take(BPC * 2);
  __bf16* hbf   = (__bf16*)take(BPF * 2);
  size_t padBytes = (size_t)((char*)take(0) - padPool);
  __bf16* vbf   = (__bf16*)take(BCT * 2);  // time-major (B,C,T), fully written
  __bf16* wpre  = (__bf16*)take(CCK * 2);  // (C, K, C)
  __bf16* wqb   = (__bf16*)take(LCC * 2);
  __bf16* wkb   = (__bf16*)take(LCC * 2);
  __bf16* wvb   = (__bf16*)take(LCC * 2);
  __bf16* wob   = (__bf16*)take(LCC * 2);
  __bf16* wf1   = (__bf16*)take(LF1 * 2);  // (L*FC, K, C)
  __bf16* wf2   = (__bf16*)take(LF2 * 2);  // (L*C, K, FC)

  hipMemsetAsync(padPool, 0, padBytes, stream);

  tdm_probe_kernel<<<1, 32, 0, stream>>>();

  cvt_bf16_kernel<<<2048, 256, 0, stream>>>(qw, wqb, LCC);
  cvt_bf16_kernel<<<2048, 256, 0, stream>>>(kw, wkb, LCC);
  cvt_bf16_kernel<<<2048, 256, 0, stream>>>(vw, wvb, LCC);
  cvt_bf16_kernel<<<2048, 256, 0, stream>>>(ow, wob, LCC);
  cvt_w_tap_kernel<<<2048, 256, 0, stream>>>(prenet_w, wpre, Cc, Cc, Kc);
  cvt_w_tap_kernel<<<2048, 256, 0, stream>>>(f1w, wf1, Lc * FCc, Cc, Kc);
  cvt_w_tap_kernel<<<2048, 256, 0, stream>>>(f2w, wf2, Lc * Cc, FCc, Kc);

  pre_kernel<<<(int)((BCT + 255) / 256), 256, 0, stream>>>(
      x, norm_f0, spk_emb, cond_w, cond_b, f0pre_w, f0pre_b, xpreb);

  dim3 gC(Tc / 64, Cc / 16, Bc);
  dim3 gF(Tc / 64, FCc / 16, Bc);

  // prenet conv (K=3, pad 1): writes xf (f32) + xbf (bf16), * mask
  convgemm_wmma<Cc, Cc, 3, 1, false, false><<<gC, 32, 0, stream>>>(
      wpre, xpreb, prenet_b, x_mask, xf, xbf);

  for (int i = 0; i < Lc; ++i) {
    const size_t woff = (size_t)i * Cc * Cc;
    convgemm_wmma<Cc, Cc, 1, 0, false, false><<<gC, 32, 0, stream>>>(
        wqb + woff, xbf, qb + i * Cc, nullptr, nullptr, qbf);
    convgemm_wmma<Cc, Cc, 1, 0, false, false><<<gC, 32, 0, stream>>>(
        wkb + woff, xbf, kb + i * Cc, nullptr, nullptr, kbf);
    convgemm_wmma<Cc, Cc, 1, 0, false, true><<<gC, 32, 0, stream>>>(
        wvb + woff, xbf, vb + i * Cc, nullptr, nullptr, vbf);  // V time-major
    attn_flash_kernel<<<dim3(Tc / 16, Hc, Bc), 32, 0, stream>>>(qbf, kbf, vbf, abf);
    convgemm_wmma<Cc, Cc, 1, 0, false, false><<<gC, 32, 0, stream>>>(
        wob + woff, abf, ob + i * Cc, nullptr, yf, nullptr);
    ln_kernel<<<dim3(Tc, Bc), 32, 0, stream>>>(xf, yf, ln0_g + i * Cc, ln0_b + i * Cc,
                                               xf, xbf);
    convgemm_wmma<FCc, Cc, 3, 2, true, false><<<gF, 32, 0, stream>>>(
        wf1 + (size_t)i * FCc * Kc * Cc, xbf, f1b + i * FCc, x_mask, nullptr, hbf);
    convgemm_wmma<Cc, FCc, 3, 2, false, false><<<gC, 32, 0, stream>>>(
        wf2 + (size_t)i * Cc * Kc * FCc, hbf, f2b + i * Cc, x_mask, yf, nullptr);
    ln_kernel<<<dim3(Tc, Bc), 32, 0, stream>>>(xf, yf, ln1_g + i * Cc, ln1_b + i * Cc,
                                               xf, xbf);
  }

  proj_kernel<<<dim3(Tc, Bc), 32, 0, stream>>>(xf, proj_w, proj_b, x_mask,
                                               (float*)d_out);
  (void)in_sizes; (void)n_in; (void)out_size; (void)ws_size;
}
